// MPNN_88880053224254
// MI455X (gfx1250) — matrix operand order repaired and checked by reading, not compiled, not run on hardware
//
#include <hip/hip_runtime.h>

#define NN 65536
#define EE 393216
#define BB 8
#define HID 128
#define TW 25
#define NLAYERS 6
#define KP 288   // padded K for 284 (message) / 258 (update)
#define KE 32    // padded K for 28 (embedding)

typedef __attribute__((ext_vector_type(16))) __bf16 v16bf;
typedef __attribute__((ext_vector_type(8)))  float  v8f;

union ABu { v16bf v; uint4 q[2]; };

__device__ __forceinline__ __bf16 to_bf16(float x) {
  unsigned u = __builtin_bit_cast(unsigned, x);
  unsigned r = (u + 0x7FFFu + ((u >> 16) & 1u)) >> 16;
  unsigned short s = (unsigned short)r;
  return __builtin_bit_cast(__bf16, s);
}

__device__ __forceinline__ float swishf(float x) { return x / (1.f + __expf(-x)); }

// A fragment from LDS tile (row-major, `stride` bf16 per row), 16 rows x K=32.
// Layout (ISA 7.12.2, 16-bit A 16x32): lane<16: row=lane, K = {k..k+7, k+16..k+23}
//                                      lane>=16: row=lane-16, K halves shifted by +8.
__device__ __forceinline__ v16bf load_a_lds(const __bf16* base, int stride, int k, int lane) {
  int row  = lane & 15;
  int koff = (lane >> 4) << 3;
  ABu t;
  t.q[0] = *(const uint4*)(base + row * stride + k + koff);
  t.q[1] = *(const uint4*)(base + row * stride + k + 16 + koff);
  return t.v;
}

// B fragment from global W^T[out][Kpad] bf16: lane<16: col=col0+lane, K = {k..k+7, k+16..k+23};
// lane>=16: col=col0+lane-16, K halves shifted by +8.
__device__ __forceinline__ v16bf load_b_g(const __bf16* wT, int Kpad, int col0, int k, int lane) {
  int col = col0 + (lane & 15);
  int kk  = k + ((lane >> 4) << 3);
  const uint4* p = (const uint4*)(wT + (size_t)col * Kpad + kk);
  ABu t;
  t.q[0] = p[0];
  t.q[1] = p[2];
  return t.v;
}

template <int KTOT>
__device__ __forceinline__ void gemm_tile(const __bf16* aBase, int aStride,
                                          const __bf16* wT, int Kpad, int lane, v8f acc[8]) {
  for (int k = 0; k < KTOT; k += 32) {
    v16bf a = load_a_lds(aBase, aStride, k, lane);
#pragma unroll
    for (int ct = 0; ct < 8; ct++) {
      v16bf b = load_b_g(wT, Kpad, ct * 16, k, lane);
      acc[ct] = __builtin_amdgcn_wmma_f32_16x16x32_bf16(false, a, false, b,
                                                        (short)0, acc[ct], false, false);
    }
  }
}

// ---------------- utility kernels ----------------

__global__ void k_zero(float* p, long n) {
  long i   = (long)blockIdx.x * blockDim.x + threadIdx.x;
  long str = (long)gridDim.x * blockDim.x;
  for (; i < n; i += str) p[i] = 0.f;
}

__global__ void k_deg(const int* __restrict__ eidx, float* __restrict__ deg) {
  int e = blockIdx.x * blockDim.x + threadIdx.x;
  if (e < EE) atomicAdd(&deg[eidx[EE + e]], 1.f);
}

__global__ void k_gcount(const int* __restrict__ batch, float* __restrict__ gct) {
  int n = blockIdx.x * blockDim.x + threadIdx.x;
  if (n < NN) atomicAdd(&gct[batch[n]], 1.f);
}

// w: [L][K][O] fp32 -> wT: [L][O][Kpad] bf16, zero padded past K
__global__ void k_transpose(const float* __restrict__ w, __bf16* __restrict__ wT,
                            int L, int K, int O, int Kpad) {
  long total = (long)L * O * Kpad;
  long idx   = (long)blockIdx.x * blockDim.x + threadIdx.x;
  if (idx >= total) return;
  int l   = (int)(idx / ((long)O * Kpad));
  int rem = (int)(idx % ((long)O * Kpad));
  int o   = rem / Kpad;
  int k   = rem % Kpad;
  float v = (k < K) ? w[(size_t)l * K * O + (size_t)k * O + o] : 0.f;
  wT[idx] = to_bf16(v);
}

// ---------------- embedding: [N,28]->128->128 ----------------

__global__ void __launch_bounds__(128)
k_embed(const float* __restrict__ u, const float* __restrict__ xp,
        const float* __restrict__ tp, const float* __restrict__ vr,
        const __bf16* __restrict__ w1T, const float* __restrict__ b1,
        const __bf16* __restrict__ w2T, const float* __restrict__ b2,
        float* __restrict__ f, __bf16* __restrict__ fbf) {
  __shared__ __align__(16) __bf16 Atile[64 * KE];
  __shared__ __align__(16) __bf16 Mtile[64 * HID];
  int tid = threadIdx.x;
  if (tid < 64) {
    int n = blockIdx.x * 64 + tid;
#pragma unroll
    for (int j = 0; j < TW; j++) Atile[tid * KE + j] = to_bf16(u[n * TW + j]);
    Atile[tid * KE + 25] = to_bf16(xp[n] * (1.f / 16.f));
    Atile[tid * KE + 26] = to_bf16(tp[n] * 0.25f);
    Atile[tid * KE + 27] = to_bf16(vr[n]);
#pragma unroll
    for (int j = 28; j < KE; j++) Atile[tid * KE + j] = to_bf16(0.f);
  }
  __syncthreads();

  int wave = tid >> 5, lane = tid & 31;
  const v8f vz = {0.f, 0.f, 0.f, 0.f, 0.f, 0.f, 0.f, 0.f};
  v8f acc[8];
#pragma unroll
  for (int ct = 0; ct < 8; ct++) acc[ct] = vz;
  gemm_tile<KE>(Atile + wave * 16 * KE, KE, w1T, KE, lane, acc);
#pragma unroll
  for (int ct = 0; ct < 8; ct++) {
    int col = ct * 16 + (lane & 15);
    float bias = b1[col];
#pragma unroll
    for (int j = 0; j < 8; j++) {
      int row = wave * 16 + ((lane < 16) ? j : j + 8);
      Mtile[row * HID + col] = to_bf16(swishf(acc[ct][j] + bias));
    }
  }
  __syncthreads();
  v8f acc2[8];
#pragma unroll
  for (int ct = 0; ct < 8; ct++) acc2[ct] = vz;
  gemm_tile<HID>(Mtile + wave * 16 * HID, HID, w2T, HID, lane, acc2);
#pragma unroll
  for (int ct = 0; ct < 8; ct++) {
    int col = ct * 16 + (lane & 15);
    float bias = b2[col];
#pragma unroll
    for (int j = 0; j < 8; j++) {
      int row = wave * 16 + ((lane < 16) ? j : j + 8);
      float x = swishf(acc2[ct][j] + bias);
      size_t idx = (size_t)(blockIdx.x * 64 + row) * HID + col;
      f[idx]   = x;
      fbf[idx] = to_bf16(x);
    }
  }
}

// ---------------- per-edge message MLP + scatter ----------------

__global__ void __launch_bounds__(128)
k_message(const __bf16* __restrict__ fbf, const float* __restrict__ u,
          const float* __restrict__ xp, const float* __restrict__ tp,
          const float* __restrict__ vr, const int* __restrict__ eidx,
          const __bf16* __restrict__ w1T, const float* __restrict__ b1,
          const __bf16* __restrict__ w2T, const float* __restrict__ b2,
          float* __restrict__ agg) {
  __shared__ __align__(16) __bf16 Atile[64 * KP];
  __shared__ __align__(16) __bf16 Mtile[64 * HID];
  __shared__ int sdst[64];

  int tid = threadIdx.x;
  int r = tid >> 1, half = tid & 1;
  int e   = blockIdx.x * 64 + r;
  int src = eidx[e];
  int dst = eidx[EE + e];
  {
    const uint4* pd = (const uint4*)(fbf + (size_t)dst * HID + half * 64);
    const uint4* ps = (const uint4*)(fbf + (size_t)src * HID + half * 64);
    uint4* qd = (uint4*)(Atile + r * KP + half * 64);
    uint4* qs = (uint4*)(Atile + r * KP + HID + half * 64);
#pragma unroll
    for (int i = 0; i < 8; i++) { qd[i] = pd[i]; qs[i] = ps[i]; }
  }
  if (half == 0) {
    sdst[r] = dst;
#pragma unroll
    for (int j = 0; j < TW; j++)
      Atile[r * KP + 256 + j] = to_bf16(u[dst * TW + j] - u[src * TW + j]);
    Atile[r * KP + 281] = to_bf16((xp[dst] - xp[src]) * (1.f / 16.f));
    Atile[r * KP + 282] = to_bf16(tp[dst] * 0.25f);
    Atile[r * KP + 283] = to_bf16(vr[dst]);
#pragma unroll
    for (int j = 284; j < KP; j++) Atile[r * KP + j] = to_bf16(0.f);
  }
  __syncthreads();

  int wave = tid >> 5, lane = tid & 31;
  const v8f vz = {0.f, 0.f, 0.f, 0.f, 0.f, 0.f, 0.f, 0.f};
  v8f acc[8];
#pragma unroll
  for (int ct = 0; ct < 8; ct++) acc[ct] = vz;
  gemm_tile<KP>(Atile + wave * 16 * KP, KP, w1T, KP, lane, acc);
#pragma unroll
  for (int ct = 0; ct < 8; ct++) {
    int col = ct * 16 + (lane & 15);
    float bias = b1[col];
#pragma unroll
    for (int j = 0; j < 8; j++) {
      int row = wave * 16 + ((lane < 16) ? j : j + 8);
      Mtile[row * HID + col] = to_bf16(swishf(acc[ct][j] + bias));
    }
  }
  __syncthreads();
  v8f acc2[8];
#pragma unroll
  for (int ct = 0; ct < 8; ct++) acc2[ct] = vz;
  gemm_tile<HID>(Mtile + wave * 16 * HID, HID, w2T, HID, lane, acc2);
#pragma unroll
  for (int ct = 0; ct < 8; ct++) {
    int col = ct * 16 + (lane & 15);
    float bias = b2[col];
#pragma unroll
    for (int j = 0; j < 8; j++) {
      int row = wave * 16 + ((lane < 16) ? j : j + 8);
      float x = swishf(acc2[ct][j] + bias);
      atomicAdd(&agg[(size_t)sdst[row] * HID + col], x);
    }
  }
}

// ---------------- node update MLP + residual + IN stats ----------------

__global__ void __launch_bounds__(128)
k_update(const float* __restrict__ f, const __bf16* __restrict__ fbf,
         const float* __restrict__ agg, const float* __restrict__ deg,
         const float* __restrict__ tp, const float* __restrict__ vr,
         const int* __restrict__ batch,
         const __bf16* __restrict__ w1T, const float* __restrict__ b1,
         const __bf16* __restrict__ w2T, const float* __restrict__ b2,
         float* __restrict__ h, float* __restrict__ ssum, float* __restrict__ ssq) {
  __shared__ __align__(16) __bf16 Atile[64 * KP];
  __shared__ __align__(16) __bf16 Mtile[64 * HID];
  __shared__ int sb[64];

  int tid = threadIdx.x;
  int r = tid >> 1, half = tid & 1;
  int n = blockIdx.x * 64 + r;
  {
    const uint4* pf = (const uint4*)(fbf + (size_t)n * HID + half * 64);
    uint4* qf = (uint4*)(Atile + r * KP + half * 64);
#pragma unroll
    for (int i = 0; i < 8; i++) qf[i] = pf[i];
    float inv = 1.f / fmaxf(deg[n], 1.f);
#pragma unroll
    for (int i = 0; i < 64; i++)
      Atile[r * KP + HID + half * 64 + i] =
          to_bf16(agg[(size_t)n * HID + half * 64 + i] * inv);
  }
  if (half == 0) {
    sb[r] = batch[n];
    Atile[r * KP + 256] = to_bf16(tp[n] * 0.25f);
    Atile[r * KP + 257] = to_bf16(vr[n]);
#pragma unroll
    for (int j = 258; j < KP; j++) Atile[r * KP + j] = to_bf16(0.f);
  }
  __syncthreads();

  int wave = tid >> 5, lane = tid & 31;
  const v8f vz = {0.f, 0.f, 0.f, 0.f, 0.f, 0.f, 0.f, 0.f};
  v8f acc[8];
#pragma unroll
  for (int ct = 0; ct < 8; ct++) acc[ct] = vz;
  gemm_tile<KP>(Atile + wave * 16 * KP, KP, w1T, KP, lane, acc);
#pragma unroll
  for (int ct = 0; ct < 8; ct++) {
    int col = ct * 16 + (lane & 15);
    float bias = b1[col];
#pragma unroll
    for (int j = 0; j < 8; j++) {
      int row = wave * 16 + ((lane < 16) ? j : j + 8);
      Mtile[row * HID + col] = to_bf16(swishf(acc[ct][j] + bias));
    }
  }
  __syncthreads();
  v8f acc2[8];
#pragma unroll
  for (int ct = 0; ct < 8; ct++) acc2[ct] = vz;
  gemm_tile<HID>(Mtile + wave * 16 * HID, HID, w2T, HID, lane, acc2);
#pragma unroll
  for (int ct = 0; ct < 8; ct++) {
    int col = ct * 16 + (lane & 15);
    float bias = b2[col];
#pragma unroll
    for (int j = 0; j < 8; j++) {
      int row = wave * 16 + ((lane < 16) ? j : j + 8);
      float x = swishf(acc2[ct][j] + bias);
      size_t idx = (size_t)(blockIdx.x * 64 + row) * HID + col;
      float hv = f[idx] + x;
      h[idx] = hv;
      int b = sb[row];
      atomicAdd(&ssum[b * HID + col], hv);
      atomicAdd(&ssq[b * HID + col], hv * hv);
    }
  }
}

__global__ void k_finalize(const float* __restrict__ ssum, const float* __restrict__ ssq,
                           const float* __restrict__ gct,
                           float* __restrict__ meanb, float* __restrict__ istdb) {
  int i = blockIdx.x * blockDim.x + threadIdx.x;
  if (i >= BB * HID) return;
  int b = i >> 7;
  float gc = fmaxf(gct[b], 1.f);
  float m = ssum[i] / gc;
  float v = ssq[i] / gc - m * m;
  meanb[i] = m;
  istdb[i] = rsqrtf(v + 1e-5f);
}

__global__ void k_norm(const float* __restrict__ h, const int* __restrict__ batch,
                       const float* __restrict__ meanb, const float* __restrict__ istdb,
                       float* __restrict__ f, __bf16* __restrict__ fbf) {
  long i = (long)blockIdx.x * blockDim.x + threadIdx.x;
  if (i >= (long)NN * HID) return;
  int n = (int)(i >> 7);
  int c = (int)(i & 127);
  int b = batch[n];
  float v = (h[i] - meanb[b * HID + c]) * istdb[b * HID + c];
  f[i]   = v;
  fbf[i] = to_bf16(v);
}

// ---------------- output conv head ----------------

__global__ void k_head(const float* __restrict__ f, const float* __restrict__ u,
                       const float* __restrict__ c1w, const float* __restrict__ c1b,
                       const float* __restrict__ c2w, const float* __restrict__ c2b,
                       float* __restrict__ out) {
  int n = blockIdx.x * blockDim.x + threadIdx.x;
  if (n >= NN) return;
  const float* fr = f + (size_t)n * HID;
  float t1[8][38];
  for (int oc = 0; oc < 8; oc++) {
    float b = c1b[oc];
    for (int p = 0; p < 38; p++) {
      float s = 0.f;
#pragma unroll
      for (int k = 0; k < 16; k++) s += fr[3 * p + k] * c1w[oc * 16 + k];
      s += b;
      t1[oc][p] = swishf(s);
    }
  }
  float ulast = u[n * TW + TW - 1];
  float b2 = c2b[0];
  for (int j = 0; j < TW; j++) {
    float s = 0.f;
    for (int ic = 0; ic < 8; ic++)
#pragma unroll
      for (int k = 0; k < 14; k++) s += t1[ic][j + k] * c2w[ic * 14 + k];
    float diff = s + b2;
    out[n * TW + j] = ulast + (4.f / 250.f) * (float)(j + 1) * diff;
  }
}

// ---------------- host launcher ----------------

extern "C" void kernel_launch(void* const* d_in, const int* in_sizes, int n_in,
                              void* d_out, int out_size, void* d_ws, size_t ws_size,
                              hipStream_t stream) {
  const float* u      = (const float*)d_in[0];
  const float* xp     = (const float*)d_in[1];
  const float* tp     = (const float*)d_in[2];
  const float* vr     = (const float*)d_in[3];
  const int*   eidx   = (const int*)d_in[4];
  const int*   batch  = (const int*)d_in[5];
  const float* emb_w1 = (const float*)d_in[6];
  const float* emb_b1 = (const float*)d_in[7];
  const float* emb_w2 = (const float*)d_in[8];
  const float* emb_b2 = (const float*)d_in[9];
  const float* m1_w   = (const float*)d_in[10];
  const float* m1_b   = (const float*)d_in[11];
  const float* m2_w   = (const float*)d_in[12];
  const float* m2_b   = (const float*)d_in[13];
  const float* u1_w   = (const float*)d_in[14];
  const float* u1_b   = (const float*)d_in[15];
  const float* u2_w   = (const float*)d_in[16];
  const float* u2_b   = (const float*)d_in[17];
  const float* c1_w   = (const float*)d_in[18];
  const float* c1_b   = (const float*)d_in[19];
  const float* c2_w   = (const float*)d_in[20];
  const float* c2_b   = (const float*)d_in[21];

  char* wsp = (char*)d_ws;
  size_t off = 0;
  auto alloc = [&](size_t bytes) -> char* {
    char* p = wsp + off;
    off += (bytes + 255) & ~(size_t)255;
    return p;
  };
  float*  f     = (float*)alloc((size_t)NN * HID * 4);
  float*  hb    = (float*)alloc((size_t)NN * HID * 4);
  float*  agg   = (float*)alloc((size_t)NN * HID * 4);
  __bf16* fbf   = (__bf16*)alloc((size_t)NN * HID * 2);
  float*  deg   = (float*)alloc((size_t)NN * 4);         // gct contiguous after deg
  float*  gct   = (float*)alloc((size_t)BB * 4);
  float*  ssum  = (float*)alloc((size_t)BB * HID * 4);   // ssq contiguous after ssum
  float*  ssq   = (float*)alloc((size_t)BB * HID * 4);
  float*  meanb = (float*)alloc((size_t)BB * HID * 4);
  float*  istdb = (float*)alloc((size_t)BB * HID * 4);
  __bf16* e1T   = (__bf16*)alloc((size_t)HID * KE * 2);
  __bf16* e2T   = (__bf16*)alloc((size_t)HID * HID * 2);
  __bf16* m1T   = (__bf16*)alloc((size_t)NLAYERS * HID * KP * 2);
  __bf16* m2T   = (__bf16*)alloc((size_t)NLAYERS * HID * HID * 2);
  __bf16* u1T   = (__bf16*)alloc((size_t)NLAYERS * HID * KP * 2);
  __bf16* u2T   = (__bf16*)alloc((size_t)NLAYERS * HID * HID * 2);
  if (off > ws_size) return;  // insufficient scratch: refuse to run out of bounds

  // degree / graph counts (deg and gct are contiguous -> one zero pass)
  k_zero<<<256, 256, 0, stream>>>(deg, (long)NN + BB);
  k_deg<<<EE / 256, 256, 0, stream>>>(eidx, deg);
  k_gcount<<<NN / 256, 256, 0, stream>>>(batch, gct);

  // weight conversion: transposed, K-padded bf16
  k_transpose<<<(HID * KE + 255) / 256, 256, 0, stream>>>(emb_w1, e1T, 1, 28, HID, KE);
  k_transpose<<<(HID * HID + 255) / 256, 256, 0, stream>>>(emb_w2, e2T, 1, HID, HID, HID);
  k_transpose<<<(NLAYERS * HID * KP + 255) / 256, 256, 0, stream>>>(m1_w, m1T, NLAYERS, 284, HID, KP);
  k_transpose<<<(NLAYERS * HID * HID + 255) / 256, 256, 0, stream>>>(m2_w, m2T, NLAYERS, HID, HID, HID);
  k_transpose<<<(NLAYERS * HID * KP + 255) / 256, 256, 0, stream>>>(u1_w, u1T, NLAYERS, 258, HID, KP);
  k_transpose<<<(NLAYERS * HID * HID + 255) / 256, 256, 0, stream>>>(u2_w, u2T, NLAYERS, HID, HID, HID);

  k_embed<<<NN / 64, 128, 0, stream>>>(u, xp, tp, vr, e1T, emb_b1, e2T, emb_b2, f, fbf);

  for (int l = 0; l < NLAYERS; l++) {
    k_zero<<<2048, 256, 0, stream>>>(agg, (long)NN * HID);
    k_zero<<<8, 256, 0, stream>>>(ssum, (long)2 * BB * HID);  // ssum+ssq contiguous
    k_message<<<EE / 64, 128, 0, stream>>>(
        fbf, u, xp, tp, vr, eidx,
        m1T + (size_t)l * HID * KP, m1_b + l * HID,
        m2T + (size_t)l * HID * HID, m2_b + l * HID, agg);
    k_update<<<NN / 64, 128, 0, stream>>>(
        f, fbf, agg, deg, tp, vr, batch,
        u1T + (size_t)l * HID * KP, u1_b + l * HID,
        u2T + (size_t)l * HID * HID, u2_b + l * HID,
        hb, ssum, ssq);
    k_finalize<<<4, 256, 0, stream>>>(ssum, ssq, gct, meanb, istdb);
    k_norm<<<(NN * HID) / 256, 256, 0, stream>>>(hb, batch, meanb, istdb, f, fbf);
  }

  k_head<<<NN / 256, 256, 0, stream>>>(f, u, c1_w, c1_b, c2_w, c2_b, (float*)d_out);
}
